// CLFModel_73899207295236
// MI455X (gfx1250) — compile-verified, matrix operand order checked by reading, not run on hardware
//
#include <hip/hip_runtime.h>
#include <hip/hip_bf16.h>

// Problem constants
#define B_   256
#define T_   512
#define H_   512
#define C_   10
#define G_   2048          // 4*H (gates i,f,g,o)
#define KB_  16            // K blocks of 32 (bf16 WMMA K=32)
#define NT_  128           // G_/16 N-tiles
#define ROW  520           // padded LDS row stride (halves): bank-conflict-free

// Packed-weight byte-offset strides (see pack_weights):
//   tile (nt, kb) lives at byte offset (nt*16 + kb)*1024 + lane*32
//   nt = gi*32 + w*2 + jt  ->  gi*524288 + w*32768 + jt*16384 + kb*1024
#define MAT_STRIDE  2097152   // G_*H_*2 bytes per matrix
#define GI_STRIDE   524288
#define JT_STRIDE   16384
#define KB_STRIDE   1024

typedef __bf16 bf16_t;
typedef bf16_t v16bf __attribute__((ext_vector_type(16)));
typedef bf16_t v8bf  __attribute__((ext_vector_type(8)));
typedef float  v8f   __attribute__((ext_vector_type(8)));

// Global-address-space views so weight loads lower to global_load_b128
// (LOADcnt only) instead of flat_load_b128 (LOADcnt+DScnt coupling).
typedef __attribute__((address_space(1))) const char  gchar;
typedef __attribute__((address_space(1))) const v16bf gv16bf;

__device__ __forceinline__ float sigmoid_f(float x) {
    return __fdividef(1.0f, 1.0f + __expf(-x));
}
__device__ __forceinline__ float tanh_f(float x) {
    float e = __expf(-2.0f * x);
    return __fdividef(1.0f - e, 1.0f + e);
}

// Two 16B LDS chunks (row stride 1040B keeps 16 lanes on distinct bank spans).
__device__ __forceinline__ v16bf load_tile_lds(const bf16_t* p) {
    union { v16bf v; v8bf h[2]; } u;
    u.h[0] = *(const v8bf*)(p);
    u.h[1] = *(const v8bf*)(p + 8);
    return u.v;
}

// Load the 8 B-tiles (one per gate-column tile this wave owns) of k-block kbv.
#define LOAD8(dst, kbv, MOFF)                                                  \
    _Pragma("unroll")                                                          \
    for (int u = 0; u < 8; ++u)                                                \
        dst[u] = *(gv16bf*)(wb + (MOFF) + (u >> 1) * GI_STRIDE +               \
                            (u & 1) * JT_STRIDE + (kbv) * KB_STRIDE);

#define WMMA8(accv, av, bsrc)                                                  \
    _Pragma("unroll")                                                          \
    for (int u = 0; u < 8; ++u)                                                \
        accv[u] = __builtin_amdgcn_wmma_f32_16x16x32_bf16(                     \
            false, av, false, bsrc[u], (short)0, accv[u], false, false);

// One GEMM pass: acc += A(16xK from LDS, base arow) * Wmat^T, with explicit
// one-k-block software pipeline (ping-pong B buffers -> ~16 outstanding
// global_load_b128 of prefetch distance in front of every WMMA group).
template <int MOFF>
__device__ __forceinline__ void gemm_pass(v8f* acc, const bf16_t* arow, gchar* wb) {
    v16bf b0[8], b1[8];
    LOAD8(b0, 0, MOFF);
#pragma unroll 1
    for (int kb2 = 0; kb2 < KB_ / 2; ++kb2) {
        v16bf av0 = load_tile_lds(arow + kb2 * 64);
        LOAD8(b1, 2 * kb2 + 1, MOFF);
        WMMA8(acc, av0, b0);
        v16bf av1 = load_tile_lds(arow + kb2 * 64 + 32);
        LOAD8(b0, (2 * kb2 + 2) & (KB_ - 1), MOFF);   // wrap: no OOB overshoot
        WMMA8(acc, av1, b1);
    }
}

// ---------------------------------------------------------------------------
// Prep 1: fp32 -> bf16 + swizzle into B-operand order:
//   wp[((nt*16 + kb)*32 + lane)*16 + h] = W[n][k]
//   n = nt*16 + (lane&15),  k = kb*32 + (lane>>4)*16 + h
// ---------------------------------------------------------------------------
__global__ void pack_weights(const float* __restrict__ whh1,
                             const float* __restrict__ wih2,
                             const float* __restrict__ whh2,
                             bf16_t* __restrict__ wp) {
    const int PER = NT_ * KB_ * 32;           // 65536 16-half vectors / matrix
    int tid = blockIdx.x * blockDim.x + threadIdx.x;
    if (tid >= 3 * PER) return;
    int mat  = tid / PER;
    int r    = tid % PER;
    int lane = r & 31;
    int kb   = (r >> 5) & (KB_ - 1);
    int nt   = r >> 9;
    const float* src = (mat == 0) ? whh1 : ((mat == 1) ? wih2 : whh2);
    bf16_t* dst = wp + (size_t)mat * ((size_t)G_ * H_) + (size_t)r * 16;
    int n  = nt * 16 + (lane & 15);
    int k0 = kb * 32 + (lane >> 4) * 16;
    const float* s = src + (size_t)n * H_ + k0;
#pragma unroll
    for (int h = 0; h < 16; ++h) dst[h] = (bf16_t)s[h];
}

// Prep 2: fused biases (b_ih + b_hh) per layer + W_ih1 column (fp32).
__global__ void prep_vec(const float* __restrict__ bih1, const float* __restrict__ bhh1,
                         const float* __restrict__ bih2, const float* __restrict__ bhh2,
                         const float* __restrict__ wih1, float* __restrict__ fws) {
    int n = blockIdx.x * blockDim.x + threadIdx.x;
    if (n < G_) {
        fws[n]          = bih1[n] + bhh1[n];
        fws[G_ + n]     = bih2[n] + bhh2[n];
        fws[2 * G_ + n] = wih1[n];
    }
}

// ---------------------------------------------------------------------------
// Persistent 2-layer LSTM. One workgroup = 16 batch rows (one WMMA M-tile),
// 16 waves split the 2048 gate columns (wave w owns hidden units [32w,32w+32)
// for BOTH layers). h1/h2 state in LDS (bf16), c1/c2 in accumulator-layout
// registers. Weights stream from L2 every step (6MB bf16 packed, fully
// L2-resident on a 192MB L2). Batch is the only cross-step-independent axis,
// so no grid-wide sync is ever needed -- 5 workgroup barriers per timestep.
// ---------------------------------------------------------------------------
__launch_bounds__(512, 1)
__global__ void lstm_persistent(const float* __restrict__ x,
                                const bf16_t* __restrict__ wp,
                                const float* __restrict__ fws,
                                const float* __restrict__ w_out,
                                const float* __restrict__ b_out,
                                float* __restrict__ out) {
    __shared__ bf16_t h1s[16 * ROW];
    __shared__ bf16_t h2s[16 * ROW];
    __shared__ float  xs[16];

    const int tid  = threadIdx.x;
    const int w    = tid >> 5;          // wave id 0..15
    const int lane = tid & 31;
    const int grp  = lane >> 4;         // half-wave (K split / M split)
    const int col  = lane & 15;         // N (or M) within tile
    const int m0   = blockIdx.x * 16;   // batch row base

    const float* bias1 = fws;
    const float* bias2 = fws + G_;
    const float* wih1  = fws + 2 * G_;

    // Per-lane weight base as a GLOBAL-addrspace pointer; every B-tile address
    // is this base + compile-time immediate (all < 8MB signed-24-bit range).
    gchar* wb = (gchar*)(unsigned long long)
        ((const char*)wp + (size_t)(w * 32768 + lane * 32));

    // Per-lane A-tile bases in LDS (include col row + half-wave K offset).
    const bf16_t* a1row = &h1s[(size_t)col * ROW + grp * 16];
    const bf16_t* a2row = &h2s[(size_t)col * ROW + grp * 16];

    // Wave-lifetime per-tile scalars (column n of tile u=(gi*2+jt)).
    float b1v[8], b2v[8], wi1v[8];
#pragma unroll
    for (int u = 0; u < 8; ++u) {
        int gi = u >> 1, jt = u & 1;
        int n = gi * 512 + w * 32 + jt * 16 + col;
        b1v[u]  = bias1[n];
        b2v[u]  = bias2[n];
        wi1v[u] = wih1[n];
    }

    for (int i = tid; i < 16 * ROW; i += 512) {
        h1s[i] = (bf16_t)0.0f;
        h2s[i] = (bf16_t)0.0f;
    }

    v8f c1[2] = {};   // cell state, accumulator layout, per jt tile
    v8f c2[2] = {};

    __syncthreads();

#pragma unroll 1
    for (int t = 0; t < T_; ++t) {
        // Opaque-ify the weight base each step: kills cross-timestep LICM so
        // the compiler cannot hoist (and spill) the invariant weight tiles.
        asm volatile("" : "+v"(wb));

        if (tid < 16) xs[tid] = x[(size_t)(m0 + tid) * T_ + t];
        __syncthreads();                               // (A) xs + prev state ready

        // ===== layer 1: gates = h1(t-1) @ Whh1^T + x_t*Wih1 + bias =====
        v8f acc[8];
#pragma unroll
        for (int u = 0; u < 8; ++u) {
            v8f a;
#pragma unroll
            for (int r = 0; r < 8; ++r) a[r] = b1v[u] + xs[r + 8 * grp] * wi1v[u];
            acc[u] = a;
        }
        gemm_pass<0>(acc, a1row, wb);
        __syncthreads();                               // (B) all done reading h1(t-1)

#pragma unroll
        for (int jt = 0; jt < 2; ++jt)
#pragma unroll
            for (int r = 0; r < 8; ++r) {
                float iv = sigmoid_f(acc[0 + jt][r]);
                float fv = sigmoid_f(acc[2 + jt][r]);
                float gv = tanh_f  (acc[4 + jt][r]);
                float ov = sigmoid_f(acc[6 + jt][r]);
                float cn = fv * c1[jt][r] + iv * gv;
                c1[jt][r] = cn;
                float hn = ov * tanh_f(cn);
                h1s[(size_t)(r + 8 * grp) * ROW + w * 32 + jt * 16 + col] = (bf16_t)hn;
            }
        __syncthreads();                               // (C) h1(t) ready

        // ===== layer 2: gates = h1(t) @ Wih2^T + h2(t-1) @ Whh2^T + bias =====
#pragma unroll
        for (int u = 0; u < 8; ++u) {
            v8f a;
#pragma unroll
            for (int r = 0; r < 8; ++r) a[r] = b2v[u];
            acc[u] = a;
        }
        gemm_pass<MAT_STRIDE>(acc, a1row, wb);         // input contribution
        gemm_pass<2 * MAT_STRIDE>(acc, a2row, wb);     // recurrent contribution
        __syncthreads();                               // (D) all done reading h2(t-1)

#pragma unroll
        for (int jt = 0; jt < 2; ++jt)
#pragma unroll
            for (int r = 0; r < 8; ++r) {
                float iv = sigmoid_f(acc[0 + jt][r]);
                float fv = sigmoid_f(acc[2 + jt][r]);
                float gv = tanh_f  (acc[4 + jt][r]);
                float ov = sigmoid_f(acc[6 + jt][r]);
                float cn = fv * c2[jt][r] + iv * gv;
                c2[jt][r] = cn;
                float hn = ov * tanh_f(cn);
                h2s[(size_t)(r + 8 * grp) * ROW + w * 32 + jt * 16 + col] = (bf16_t)hn;
            }
        __syncthreads();                               // (E) h2(t) ready
    }

    // ===== output projection: out = h2(T-1) @ W_out^T + b_out  (C=10) =====
    for (int idx = tid; idx < 16 * C_; idx += 512) {
        int m = idx / C_, c = idx % C_;
        float s = b_out[c];
        for (int j = 0; j < H_; ++j)
            s += (float)h2s[(size_t)m * ROW + j] * w_out[c * H_ + j];
        out[(size_t)(m0 + m) * C_ + c] = s;
    }
}

// ---------------------------------------------------------------------------
extern "C" void kernel_launch(void* const* d_in, const int* in_sizes, int n_in,
                              void* d_out, int out_size, void* d_ws, size_t ws_size,
                              hipStream_t stream) {
    const float* x     = (const float*)d_in[0];
    const float* W_ih1 = (const float*)d_in[1];
    const float* W_hh1 = (const float*)d_in[2];
    const float* b_ih1 = (const float*)d_in[3];
    const float* b_hh1 = (const float*)d_in[4];
    const float* W_ih2 = (const float*)d_in[5];
    const float* W_hh2 = (const float*)d_in[6];
    const float* b_ih2 = (const float*)d_in[7];
    const float* b_hh2 = (const float*)d_in[8];
    const float* W_out = (const float*)d_in[9];
    const float* b_out = (const float*)d_in[10];
    float* out = (float*)d_out;

    bf16_t* wpack = (bf16_t*)d_ws;                                  // 6 MB bf16
    float*  fws   = (float*)((char*)d_ws + (size_t)3 * G_ * H_ * sizeof(bf16_t));

    int packN = 3 * NT_ * KB_ * 32;
    pack_weights<<<(packN + 255) / 256, 256, 0, stream>>>(W_hh1, W_ih2, W_hh2, wpack);
    prep_vec<<<(G_ + 255) / 256, 256, 0, stream>>>(b_ih1, b_hh1, b_ih2, b_hh2, W_ih1, fws);
    lstm_persistent<<<B_ / 16, 512, 0, stream>>>(x, wpack, fws, W_out, b_out, out);
}
